// VectorQuantizer_7404523618486
// MI455X (gfx1250) — matrix operand order repaired and checked by reading, not compiled, not run on hardware
//
#include <hip/hip_runtime.h>
#include <hip/hip_bf16.h>

typedef __attribute__((ext_vector_type(16))) __bf16 v16bf;
typedef __attribute__((ext_vector_type(8)))  float  v8f;

#define K_CODES 512
#define DIM 64
#define ROWS_PER_CHUNK 512   // 8 waves * 4 row-tiles * 16 rows
#define NT (K_CODES / 16)    // 32 code tiles
#define BIG 64.0f            // positivity bias folded into C init

__device__ __forceinline__ unsigned short f2bf(float f) {
  unsigned u = __float_as_uint(f);
  u += 0x7FFFu + ((u >> 16) & 1u);   // round-to-nearest-even
  return (unsigned short)(u >> 16);
}

union V16U { v16bf v; unsigned short u[16]; };

__global__ void __launch_bounds__(256) vq_main(
    const float* __restrict__ inp, const float* __restrict__ emb,
    float* __restrict__ out, float* __restrict__ partial, int nChunks)
{
  __shared__ __align__(32) unsigned short s_emb[K_CODES * DIM]; // bf16(-2*emb)
  __shared__ float s_e2[K_CODES];                               // ||e||^2 (f32)
  __shared__ int   s_idx[ROWS_PER_CHUNK];

  const int tid  = threadIdx.x;
  const int lane = tid & 31;
  const int wv   = tid >> 5;     // 0..7
  const int lm   = lane & 15;    // position within 16-lane half
  const int hf   = lane >> 4;    // half select

  // ---- Setup: codebook -> LDS (bf16, scaled by -2) + ||e||^2 ----
  for (int c = tid; c < K_CODES; c += 256) {
    const float4* er = (const float4*)(emb + (size_t)c * DIM);
    unsigned short* dst = s_emb + c * DIM;
    float ss = 0.f;
    #pragma unroll
    for (int q = 0; q < DIM / 4; ++q) {
      float4 v = er[q];
      ss += v.x * v.x + v.y * v.y + v.z * v.z + v.w * v.w;
      dst[q * 4 + 0] = f2bf(-2.f * v.x);
      dst[q * 4 + 1] = f2bf(-2.f * v.y);
      dst[q * 4 + 2] = f2bf(-2.f * v.z);
      dst[q * 4 + 3] = f2bf(-2.f * v.w);
    }
    s_e2[c] = ss;
  }
  __syncthreads();

  // per-lane (||e||^2 + BIG) for codes (nt*16 + lm); BIG rides for free in C init
  float e2reg[NT];
  #pragma unroll
  for (int nt = 0; nt < NT; ++nt) e2reg[nt] = s_e2[nt * 16 + lm] + BIG;

  float total = 0.f;

  for (int chunk = blockIdx.x; chunk < nChunks; chunk += gridDim.x) {
    const size_t rowBase = (size_t)chunk * ROWS_PER_CHUNK;

    // ---- Phase A: build A fragments (4 row tiles x K-steps {0..31, 32..63}) ----
    v16bf a[4][2];
    #pragma unroll
    for (int t = 0; t < 4; ++t) {
      const size_t r = rowBase + (size_t)wv * 64 + t * 16 + lm;
      const float* rp = inp + r * DIM;
      #pragma unroll
      for (int s = 0; s < 2; ++s) {
        // 16-bit A layout (contiguous, matches B): elems 0..15 -> K = 32s+16h..32s+16h+15
        const float* p0 = rp + 32 * s + 16 * hf;
        const float* p1 = p0 + 8;
        float4 c00 = *(const float4*)(p0);
        float4 c01 = *(const float4*)(p0 + 4);
        float4 c10 = *(const float4*)(p1);
        float4 c11 = *(const float4*)(p1 + 4);
        V16U uu;
        uu.u[0]  = f2bf(c00.x); uu.u[1]  = f2bf(c00.y); uu.u[2]  = f2bf(c00.z); uu.u[3]  = f2bf(c00.w);
        uu.u[4]  = f2bf(c01.x); uu.u[5]  = f2bf(c01.y); uu.u[6]  = f2bf(c01.z); uu.u[7]  = f2bf(c01.w);
        uu.u[8]  = f2bf(c10.x); uu.u[9]  = f2bf(c10.y); uu.u[10] = f2bf(c10.z); uu.u[11] = f2bf(c10.w);
        uu.u[12] = f2bf(c11.x); uu.u[13] = f2bf(c11.y); uu.u[14] = f2bf(c11.z); uu.u[15] = f2bf(c11.w);
        a[t][s] = uu.v;
      }
    }

    // packed (score | code) keys: positive-float bits are u32-monotonic,
    // low 9 mantissa bits replaced by code -> ties resolve to smaller index
    unsigned keys[4][8];
    #pragma unroll
    for (int t = 0; t < 4; ++t)
      #pragma unroll
      for (int j = 0; j < 8; ++j) keys[t][j] = 0xFFFFFFFFu;

    // ---- sweep all 512 codes: score = ||e||^2 + BIG + A x (-2E) ----
    for (int nt = 0; nt < NT; ++nt) {
      const unsigned short* bp = s_emb + (nt * 16 + lm) * DIM + hf * 16;
      v16bf b0 = *(const v16bf*)(bp);        // K = 0..31 block (16 contiguous per half)
      v16bf b1 = *(const v16bf*)(bp + 32);   // K = 32..63 block
      const float e2v = e2reg[nt];
      const unsigned code = (unsigned)(nt * 16 + lm);
      #pragma unroll
      for (int t = 0; t < 4; ++t) {
        v8f c = { e2v, e2v, e2v, e2v, e2v, e2v, e2v, e2v };
        c = __builtin_amdgcn_wmma_f32_16x16x32_bf16(false, a[t][0], false, b0, (short)0, c, false, false);
        c = __builtin_amdgcn_wmma_f32_16x16x32_bf16(false, a[t][1], false, b1, (short)0, c, false, false);
        #pragma unroll
        for (int j = 0; j < 8; ++j) {
          unsigned key = (__float_as_uint(c[j]) & 0xFFFFFE00u) | code; // v_and_or_b32
          unsigned cur = keys[t][j];
          keys[t][j] = key < cur ? key : cur;                          // v_min_u32
        }
      }
    }

    // ---- argmin across the 16-lane n-dimension (stays within halves) ----
    #pragma unroll
    for (int t = 0; t < 4; ++t) {
      #pragma unroll
      for (int j = 0; j < 8; ++j) {
        unsigned k = keys[t][j];
        #pragma unroll
        for (int m = 1; m <= 8; m <<= 1) {
          unsigned o = (unsigned)__shfl_xor((int)k, m, 32);
          k = o < k ? o : k;
        }
        if (lm == 0) s_idx[wv * 64 + t * 16 + hf * 8 + j] = (int)(k & 511u); // row m = j + 8*half
      }
    }
    __syncthreads();

    // ---- Phase B: gather f32 rows, write output, accumulate (q - x)^2 ----
    const size_t base4 = rowBase * (DIM / 4);
    #pragma unroll 4
    for (int it = 0; it < (ROWS_PER_CHUNK * DIM / 4) / 256; ++it) {
      int li   = it * 256 + tid;   // local float4 index
      int lrow = li >> 4;
      int d4   = li & 15;
      int code = s_idx[lrow];
      float4 e4 = *(const float4*)(emb + (size_t)code * DIM + d4 * 4);
      float4 x4 = *(const float4*)(inp + (base4 + li) * 4);
      ((float4*)out)[base4 + li] = e4;
      float dx = e4.x - x4.x, dy = e4.y - x4.y, dz = e4.z - x4.z, dw = e4.w - x4.w;
      total += dx * dx + dy * dy + dz * dz + dw * dw;
    }
    __syncthreads();  // protect s_idx before next chunk
  }

  // ---- deterministic block reduction of loss partials ----
  #pragma unroll
  for (int m = 16; m >= 1; m >>= 1) total += __shfl_xor(total, m, 32);
  if (lane == 0) s_e2[wv] = total;   // s_e2 reuse (e2reg already in registers)
  __syncthreads();
  if (tid == 0) {
    float s = 0.f;
    for (int w = 0; w < 8; ++w) s += s_e2[w];
    partial[blockIdx.x] = s;
  }
}

__global__ void vq_finalize(const float* __restrict__ partial, int n,
                            float* __restrict__ out, long long totalElems)
{
  __shared__ float sh[256];
  float s = 0.f;
  for (int i = threadIdx.x; i < n; i += 256) s += partial[i];
  sh[threadIdx.x] = s;
  __syncthreads();
  for (int st = 128; st > 0; st >>= 1) {
    if (threadIdx.x < st) sh[threadIdx.x] += sh[threadIdx.x + st];
    __syncthreads();
  }
  if (threadIdx.x == 0) {
    float mean = sh[0] / (float)totalElems;
    out[totalElems]     = 0.25f * mean;  // commitment_loss
    out[totalElems + 1] = mean;          // embedding_loss
  }
}

extern "C" void kernel_launch(void* const* d_in, const int* in_sizes, int n_in,
                              void* d_out, int out_size, void* d_ws, size_t ws_size,
                              hipStream_t stream)
{
  const float* inp = (const float*)d_in[0];
  const float* emb = (const float*)d_in[1];
  float* out = (float*)d_out;

  const long long nElems = (long long)in_sizes[0];       // 33 554 432
  const long long rows   = nElems / DIM;                 // 524 288
  int chunks = (int)(rows / ROWS_PER_CHUNK);             // 1024
  int blocks = chunks < 512 ? chunks : 512;
  if (blocks < 1) blocks = 1;

  float* partial = (float*)d_ws;                         // blocks * 4 bytes

  vq_main<<<blocks, 256, 0, stream>>>(inp, emb, out, partial, chunks);
  vq_finalize<<<1, 256, 0, stream>>>(partial, blocks, out, nElems);
}
